// Discriminator_2491081032169
// MI455X (gfx1250) — matrix operand order repaired and checked by reading, not compiled, not run on hardware
//
#include <hip/hip_runtime.h>

typedef __attribute__((ext_vector_type(16))) __bf16 v16bf;
typedef __attribute__((ext_vector_type(8)))  float  v8f;

#define D_IN  1024
#define D_HID 256

// ---------------------------------------------------------------- degrees
__global__ void deg_kernel(const int* __restrict__ src, const int* __restrict__ dst,
                           unsigned* __restrict__ dout, unsigned* __restrict__ din, int E) {
  int e = blockIdx.x * 256 + threadIdx.x;
  if (e < E) {
    atomicAdd(&dout[src[e]], 1u);
    atomicAdd(&din[dst[e]], 1u);
  }
}

__global__ void norm_kernel(const unsigned* __restrict__ dout, const unsigned* __restrict__ din,
                            float* __restrict__ ns, float* __restrict__ nd, int N) {
  int n = blockIdx.x * 256 + threadIdx.x;
  if (n < N) {
    ns[n] = rsqrtf(fmaxf((float)dout[n], 1.0f));
    nd[n] = rsqrtf(fmaxf((float)din[n], 1.0f));
  }
}

// ------------------------------------------------- pack W1 into WMMA B layout
// Element (k, n) of W1 (k-major, row stride 256) goes to
//   off = ((k>>5)*256 + n)*32 + (k&31)
// so a lane's 16 B-fragment values (16 consecutive K at half*16) are one
// contiguous 32-byte run -> two global_load_b128 per fragment.
__global__ void packw_kernel(const float* __restrict__ W, __bf16* __restrict__ hi,
                             __bf16* __restrict__ lo) {
  int idx = blockIdx.x * 256 + threadIdx.x;
  if (idx < D_IN * D_HID) {
    int k = idx >> 8, n = idx & 255;
    float v = W[idx];
    __bf16 h = (__bf16)v;
    float r = v - (float)h;
    size_t o = ((size_t)((k >> 5) * D_HID + n)) * 32 + (size_t)(k & 31);
    hi[o] = h;
    lo[o] = (__bf16)r;
  }
}

// ----------------------------------------------- GEMM1: h = (x*ns) @ W1
// block = 256 threads = 8 waves in 2 (rows) x 4 (cols); wave tile 16x64.
// Split-bf16: acc += Ahi*Blo + Alo*Bhi + Ahi*Bhi  (3 WMMAs per 16x16x32 step)
__global__ void __launch_bounds__(256) gemm1_kernel(
    const float* __restrict__ x, const float* __restrict__ ns,
    const __bf16* __restrict__ whi, const __bf16* __restrict__ wlo,
    float* __restrict__ h, int N) {
  const int lane = threadIdx.x & 31;
  const int wave = threadIdx.x >> 5;
  const int wr = wave >> 2;   // 0..1 : row tile
  const int wc = wave & 3;    // 0..3 : 64-col strip
  const int half = lane >> 4; // lane group 0/1
  const int rowBase = blockIdx.x * 32 + wr * 16;

  int row = rowBase + (lane & 15);
  if (row > N - 1) row = N - 1;          // clamp loads; keep EXEC all-ones for WMMA
  const float nrm = ns[row];
  const float* __restrict__ xrow = x + (size_t)row * D_IN;

  v8f acc0 = {}, acc1 = {}, acc2 = {}, acc3 = {};

  for (int kt = 0; kt < D_IN / 32; ++kt) {
    // A fragment: lane holds K = half*16 + [0..15] (matches packed B order):
    // one contiguous 16-float run.
    const int kb = kt * 32 + half * 16;
    // speculative prefetch ~8 k-tiles (1KB) ahead of this row's stream
    __builtin_prefetch(xrow + kb + 256, 0, 0);
    float4 a0 = *(const float4*)(xrow + kb);
    float4 a1 = *(const float4*)(xrow + kb + 4);
    float4 a2 = *(const float4*)(xrow + kb + 8);
    float4 a3 = *(const float4*)(xrow + kb + 12);
    float av[16] = {a0.x, a0.y, a0.z, a0.w, a1.x, a1.y, a1.z, a1.w,
                    a2.x, a2.y, a2.z, a2.w, a3.x, a3.y, a3.z, a3.w};
    v16bf ahi, alo;
#pragma unroll
    for (int i = 0; i < 16; ++i) {
      float f = av[i] * nrm;
      __bf16 hv = (__bf16)f;
      ahi[i] = hv;
      alo[i] = (__bf16)(f - (float)hv);
    }

    const size_t bbase = ((size_t)kt * D_HID) * 32 + (size_t)half * 16;
#pragma unroll
    for (int nt = 0; nt < 4; ++nt) {
      const int col = wc * 64 + nt * 16 + (lane & 15);
      const size_t bo = bbase + (size_t)col * 32;
      v16bf bhi = *(const v16bf*)(whi + bo);
      v16bf blo = *(const v16bf*)(wlo + bo);
      v8f a = nt == 0 ? acc0 : nt == 1 ? acc1 : nt == 2 ? acc2 : acc3;
      a = __builtin_amdgcn_wmma_f32_16x16x32_bf16(false, ahi, false, blo, (short)0, a, false, false);
      a = __builtin_amdgcn_wmma_f32_16x16x32_bf16(false, alo, false, bhi, (short)0, a, false, false);
      a = __builtin_amdgcn_wmma_f32_16x16x32_bf16(false, ahi, false, bhi, (short)0, a, false, false);
      if (nt == 0) acc0 = a; else if (nt == 1) acc1 = a; else if (nt == 2) acc2 = a; else acc3 = a;
    }
  }

  // C/D 16x16 f32 layout: VGPR r -> M = r + half*8, N = lane&15
#pragma unroll
  for (int nt = 0; nt < 4; ++nt) {
    const v8f a = nt == 0 ? acc0 : nt == 1 ? acc1 : nt == 2 ? acc2 : acc3;
    const int col = wc * 64 + nt * 16 + (lane & 15);
#pragma unroll
    for (int r = 0; r < 8; ++r) {
      int m = rowBase + r + half * 8;
      if (m < N) h[(size_t)m * D_HID + col] = a[r];
    }
  }
}

// -------------------------------------- layer-1 edge scatter: agg[dst] += h[src]
// one edge per wave: src/dst are wave-uniform (scalar-load friendly);
// each lane moves two float4 chunks (lane*4 and lane*4+128).
__global__ void scatter1_kernel(const float* __restrict__ h, const int* __restrict__ src,
                                const int* __restrict__ dst, float* __restrict__ agg, int E) {
  int e = blockIdx.x * 8 + (threadIdx.x >> 5);
  if (e >= E) return;
  int lane = threadIdx.x & 31;
  int s = src[e], d = dst[e];
  const float* __restrict__ hs = h + (size_t)s * D_HID;
  float* __restrict__ ad = agg + (size_t)d * D_HID;
#pragma unroll
  for (int half = 0; half < 2; ++half) {
    int f = lane * 4 + half * 128;
    float4 v = *(const float4*)(hs + f);
    atomicAdd(ad + f + 0, v.x);
    atomicAdd(ad + f + 1, v.y);
    atomicAdd(ad + f + 2, v.z);
    atomicAdd(ad + f + 3, v.w);
  }
}

// ---- fused: h1 = relu(agg*nd + b1); p = ns * (h1 @ W2)  (one wave per node)
__global__ void fuse_kernel(const float* __restrict__ agg, const float* __restrict__ nd,
                            const float* __restrict__ ns, const float* __restrict__ b1,
                            const float* __restrict__ W2, float* __restrict__ p, int N) {
  int n = (blockIdx.x * 256 + threadIdx.x) >> 5;
  int lane = threadIdx.x & 31;
  if (n >= N) return;
  float d = nd[n];
  float sum = 0.0f;
#pragma unroll
  for (int i = 0; i < 8; ++i) {
    int f = lane + i * 32;
    float v = fmaxf(agg[(size_t)n * D_HID + f] * d + b1[f], 0.0f);
    sum += v * W2[f];
  }
#pragma unroll
  for (int off = 16; off > 0; off >>= 1) sum += __shfl_down(sum, off, 32);
  if (lane == 0) p[n] = sum * ns[n];
}

// -------------------------------------- layer-2 edge scatter (scalar features)
__global__ void scatter2_kernel(const float* __restrict__ p, const int* __restrict__ src,
                                const int* __restrict__ dst, float* __restrict__ oagg, int E) {
  int e = blockIdx.x * 256 + threadIdx.x;
  if (e < E) atomicAdd(&oagg[dst[e]], p[src[e]]);
}

__global__ void final_kernel(const float* __restrict__ oagg, const float* __restrict__ nd,
                             const float* __restrict__ b2, float* __restrict__ out, int N) {
  int n = blockIdx.x * 256 + threadIdx.x;
  if (n < N) out[n] = fmaxf(oagg[n] * nd[n] + b2[0], 0.0f);
}

extern "C" void kernel_launch(void* const* d_in, const int* in_sizes, int n_in,
                              void* d_out, int out_size, void* d_ws, size_t ws_size,
                              hipStream_t stream) {
  (void)n_in; (void)out_size; (void)ws_size;
  const float* x  = (const float*)d_in[0];
  const int*   src = (const int*)d_in[1];
  const int*   dst = (const int*)d_in[2];
  const float* W1 = (const float*)d_in[3];
  const float* b1 = (const float*)d_in[4];
  const float* W2 = (const float*)d_in[5];
  const float* b2 = (const float*)d_in[6];

  const int E = in_sizes[1];
  const int N = in_sizes[0] / D_IN;

  // bump allocator over workspace
  char* ws = (char*)d_ws;
  size_t off = 0;
  auto alloc = [&](size_t bytes) -> char* {
    off = (off + 255) & ~(size_t)255;
    char* ptr = ws + off;
    off += bytes;
    return ptr;
  };
  unsigned* deg  = (unsigned*)alloc((size_t)2 * N * 4);
  unsigned* dout = deg;
  unsigned* din  = deg + N;
  float*  nsrm = (float*)alloc((size_t)N * 4);
  float*  ndrm = (float*)alloc((size_t)N * 4);
  __bf16* whi  = (__bf16*)alloc((size_t)D_IN * D_HID * 2);
  __bf16* wlo  = (__bf16*)alloc((size_t)D_IN * D_HID * 2);
  float*  h    = (float*)alloc((size_t)N * D_HID * 4);
  float*  agg  = (float*)alloc((size_t)N * D_HID * 4);
  float*  p    = (float*)alloc((size_t)N * 4);
  float*  oagg = (float*)alloc((size_t)N * 4);

  hipMemsetAsync(deg, 0, (size_t)2 * N * 4, stream);
  hipMemsetAsync(agg, 0, (size_t)N * D_HID * 4, stream);
  hipMemsetAsync(oagg, 0, (size_t)N * 4, stream);

  deg_kernel<<<(E + 255) / 256, 256, 0, stream>>>(src, dst, dout, din, E);
  norm_kernel<<<(N + 255) / 256, 256, 0, stream>>>(dout, din, nsrm, ndrm, N);
  packw_kernel<<<(D_IN * D_HID + 255) / 256, 256, 0, stream>>>(W1, whi, wlo);
  gemm1_kernel<<<(N + 31) / 32, 256, 0, stream>>>(x, nsrm, whi, wlo, h, N);
  scatter1_kernel<<<(E + 7) / 8, 256, 0, stream>>>(h, src, dst, agg, E);
  fuse_kernel<<<((size_t)N * 32 + 255) / 256, 256, 0, stream>>>(agg, ndrm, nsrm, b1, W2, p, N);
  scatter2_kernel<<<(E + 255) / 256, 256, 0, stream>>>(p, src, dst, oagg, E);
  final_kernel<<<(N + 255) / 256, 256, 0, stream>>>(oagg, ndrm, b2, (float*)d_out, N);
}
